// ScatterRouter_36679020708157
// MI455X (gfx1250) — matrix operand order repaired and checked by reading, not compiled, not run on hardware
//
#include <hip/hip_runtime.h>
#include <hip/hip_fp16.h>

typedef __attribute__((ext_vector_type(16))) _Float16 v16h;
typedef __attribute__((ext_vector_type(8)))  float    v8f;
typedef __attribute__((ext_vector_type(4)))  float    f4;

#define T_TOK 4096
#define DIM   2048
#define NE    8

// ---------------------------------------------------------------------------
// Kernel 1: gating GEMM (v_wmma_f32_16x16x32_f16) + softmax + top-2.
// 256 threads = 8 waves; each wave computes a 16-token x 16-col logits tile
// (E=8 zero-padded to 16). Wg^T staged in LDS as f16 once per block.
// Padding columns read a zeroed 32B LDS stub with stride 0 -> branchless,
// EXEC stays all-1s through the WMMA loop.
// ---------------------------------------------------------------------------
__global__ __launch_bounds__(256) void k_gate(const float* __restrict__ x,
                                              const float* __restrict__ Wg,
                                              int* __restrict__ top2e,
                                              float* __restrict__ top2g)
{
    __shared__ _Float16 ldsW[NE * DIM];     // Wg transposed [n][k], 32 KB
    __shared__ _Float16 ldsZ[16];           // zero stub for padded B columns
    __shared__ float    ldsL[8][16][16];    // per-wave logits staging, 8 KB

    const int tid = threadIdx.x;
    for (int i = tid; i < NE * DIM; i += 256) {
        int n = i >> 11;            // i / DIM
        int k = i & (DIM - 1);
        ldsW[i] = (_Float16)Wg[k * NE + n];
    }
    if (tid < 16) ldsZ[tid] = (_Float16)0.0f;
    __syncthreads();

    const int wave  = tid >> 5;
    const int lane  = tid & 31;
    const int tile  = blockIdx.x * 8 + wave;   // 16-token tile
    const int half  = lane >> 4;
    const int abase = 16 * half;               // A-frag K group (contiguous, matches B)
    const int ncol  = lane & 15;               // B column
    const int kbase = 16 * half;               // B-frag K group

    const float* xrow = x + (size_t)(tile * 16 + (lane & 15)) * DIM;

    // branchless B pointer: padded columns spin on the zero stub
    const _Float16* bp = (ncol < NE) ? (ldsW + ncol * DIM + kbase) : ldsZ;
    const int bstride  = (ncol < NE) ? 32 : 0;

    v8f c = {};
    for (int k0 = 0; k0 < DIM; k0 += 32) {
        const float* pa = xrow + k0 + abase;
        f4 f0 = *(const f4*)(pa);
        f4 f1 = *(const f4*)(pa + 4);
        f4 f2 = *(const f4*)(pa + 8);
        f4 f3 = *(const f4*)(pa + 12);
        v16h a;
        a[0]=(_Float16)f0.x;  a[1]=(_Float16)f0.y;  a[2]=(_Float16)f0.z;  a[3]=(_Float16)f0.w;
        a[4]=(_Float16)f1.x;  a[5]=(_Float16)f1.y;  a[6]=(_Float16)f1.z;  a[7]=(_Float16)f1.w;
        a[8]=(_Float16)f2.x;  a[9]=(_Float16)f2.y;  a[10]=(_Float16)f2.z; a[11]=(_Float16)f2.w;
        a[12]=(_Float16)f3.x; a[13]=(_Float16)f3.y; a[14]=(_Float16)f3.z; a[15]=(_Float16)f3.w;

        v16h b = *(const v16h*)bp;
        bp += bstride;

        c = __builtin_amdgcn_wmma_f32_16x16x32_f16(false, a, false, b,
                                                   (short)0, c, false, false);
    }

    // C/D layout: vgpr j, lane half h -> M = 8h + j, N = lane&15
    #pragma unroll
    for (int j = 0; j < 8; ++j)
        ldsL[wave][8 * half + j][ncol] = c[j];
    __syncthreads();

    if (lane < 16) {
        const int t = tile * 16 + lane;
        float L[NE];
        float m = -1e30f;
        #pragma unroll
        for (int e = 0; e < NE; ++e) { L[e] = ldsL[wave][lane][e]; m = fmaxf(m, L[e]); }
        float s = 0.f;
        #pragma unroll
        for (int e = 0; e < NE; ++e) { L[e] = __expf(L[e] - m); s += L[e]; }
        const float inv = 1.f / s;
        int e0 = 0; float g0 = L[0];
        #pragma unroll
        for (int e = 1; e < NE; ++e) if (L[e] > g0) { g0 = L[e]; e0 = e; }
        int e1 = (e0 == 0) ? 1 : 0; float g1 = L[e1];
        #pragma unroll
        for (int e = 0; e < NE; ++e) if (e != e0 && L[e] > g1) { g1 = L[e]; e1 = e; }
        top2e[2 * t + 0] = e0;        top2e[2 * t + 1] = e1;
        top2g[2 * t + 0] = g0 * inv;  top2g[2 * t + 1] = g1 * inv;
    }
}

// ---------------------------------------------------------------------------
// Kernel 2: per-expert exclusive prefix over token order (wave32 ballot scan).
// 8 waves = 8 experts; writes pos per (token,slot), tags, loads.
// ---------------------------------------------------------------------------
__global__ __launch_bounds__(256) void k_route(const int* __restrict__ top2e,
                                               int* __restrict__ pos,
                                               int* __restrict__ tags,
                                               int* __restrict__ loads)
{
    const int e    = threadIdx.x >> 5;        // expert id
    const int lane = threadIdx.x & 31;
    const unsigned ltmask = (1u << lane) - 1u;
    int running = 0;
    for (int cblk = 0; cblk < T_TOK / 32; ++cblk) {
        const int t = cblk * 32 + lane;
        const int e0 = top2e[2 * t], e1 = top2e[2 * t + 1];
        const bool m0 = (e0 == e), m1 = (e1 == e);
        const bool m  = m0 | m1;
        const unsigned bal = __builtin_amdgcn_ballot_w32(m);
        if (m) {
            const int p = running + __builtin_popcount(bal & ltmask);
            pos[2 * t + (m0 ? 0 : 1)] = p;
            tags[e * T_TOK + p] = t;
        }
        running += __builtin_popcount(bal);
    }
    if (lane == 0) loads[e] = running;
}

// ---------------------------------------------------------------------------
// Init / zero-fill kernels
// ---------------------------------------------------------------------------
__global__ __launch_bounds__(256) void k_tags_init(int* __restrict__ tags)
{
    tags[blockIdx.x * 256 + threadIdx.x] = -1;
}

// Zero only rows that the scatter will NOT overwrite (p >= loads[e]).
__global__ __launch_bounds__(256) void k_zero_tail(float* __restrict__ outd,
                                                   const int* __restrict__ loads)
{
    const int e = blockIdx.x / T_TOK;
    const int r = blockIdx.x - e * T_TOK;
    if (r < loads[e]) return;                 // will be fully written by scatter
    f4* dst = (f4*)(outd + ((size_t)e * T_TOK + r) * DIM);
    const f4 z = {};
    #pragma unroll
    for (int i = 0; i < 2; ++i)
        dst[threadIdx.x + 256 * i] = z;
}

// ---------------------------------------------------------------------------
// Kernel 4: scaled row scatter. One block per (token, slot): 2048 floats b128.
// ---------------------------------------------------------------------------
__global__ __launch_bounds__(256) void k_scatter(const float* __restrict__ x,
                                                 const int* __restrict__ top2e,
                                                 const float* __restrict__ top2g,
                                                 const int* __restrict__ pos,
                                                 float* __restrict__ outd)
{
    const int slot = blockIdx.x;              // 0 .. 2T-1
    const int t    = slot >> 1;
    const int e    = top2e[slot];
    const float g  = top2g[slot];
    const int p    = pos[slot];
    const f4* src = (const f4*)(x + (size_t)t * DIM);
    f4* dst = (f4*)(outd + ((size_t)e * T_TOK + p) * DIM);
    #pragma unroll
    for (int i = 0; i < 2; ++i) {
        const int idx = threadIdx.x + 256 * i;
        f4 v = src[idx];
        v.x *= g; v.y *= g; v.z *= g; v.w *= g;
        dst[idx] = v;
    }
}

// ---------------------------------------------------------------------------
extern "C" void kernel_launch(void* const* d_in, const int* in_sizes, int n_in,
                              void* d_out, int out_size, void* d_ws, size_t ws_size,
                              hipStream_t stream)
{
    const float* x  = (const float*)d_in[0];
    const float* Wg = (const float*)d_in[1];

    float* out_data = (float*)d_out;
    int*   out_tags = (int*)(out_data + (size_t)NE * T_TOK * DIM);
    int*   out_loads = out_tags + NE * T_TOK;

    char* ws = (char*)d_ws;
    int*   top2e = (int*)ws;                                   // 2T ints
    float* top2g = (float*)(ws + 2 * T_TOK * sizeof(int));     // 2T floats
    int*   pos   = (int*)(ws + 4 * T_TOK * sizeof(int));       // 2T ints

    k_tags_init<<<(NE * T_TOK) / 256, 256, 0, stream>>>(out_tags);
    k_gate<<<T_TOK / 128, 256, 0, stream>>>(x, Wg, top2e, top2g);
    k_route<<<1, 256, 0, stream>>>(top2e, pos, out_tags, out_loads);
    k_zero_tail<<<NE * T_TOK, 256, 0, stream>>>(out_data, out_loads);
    k_scatter<<<2 * T_TOK, 256, 0, stream>>>(x, top2e, top2g, pos, out_data);
}
